// BEVCrossAttnTransform_33878702031095
// MI455X (gfx1250) — matrix operand order repaired and checked by reading, not compiled, not run on hardware
//
#include <hip/hip_runtime.h>
#include <hip/hip_bf16.h>

// ---------------------------------------------------------------------------
// BEV cross-attention transform for MI455X (gfx1250), wave32 + WMMA bf16.
// ---------------------------------------------------------------------------

typedef __attribute__((ext_vector_type(16))) __bf16 bf16x16;
typedef __attribute__((ext_vector_type(8)))  __bf16 bf16x8;
typedef __attribute__((ext_vector_type(4)))  __bf16 bf16x4;
typedef __attribute__((ext_vector_type(8)))  float  f32x8;

#define NQ     10000
#define CDIM   256
#define NHEADS 8
#define HD     32
#define LMAPS  12
#define NPTS   4
#define KDEP   3
#define TTOT   21120   // 6*32*88 + 6*16*44

__device__ __forceinline__ f32x8 wmma_bf16f32(bf16x16 a, bf16x16 b, f32x8 c) {
  // D = A(16x32 bf16) * B(32x16 bf16) + C(16x16 f32)
  return __builtin_amdgcn_wmma_f32_16x16x32_bf16(false, a, false, b,
                                                 (short)0, c, false, false);
}

// merge two 16-byte LDS strips into one 16-element fragment (2x ds_load_b128)
__device__ __forceinline__ bf16x16 ld_frag(const __bf16* p0, const __bf16* p1) {
  bf16x8 lo = *(const bf16x8*)p0;
  bf16x8 hi = *(const bf16x8*)p1;
  return __builtin_shufflevector(lo, hi, 0, 1, 2, 3, 4, 5, 6, 7,
                                 8, 9, 10, 11, 12, 13, 14, 15);
}

// per-map geometry (level-major: 6x(32,88) then 6x(16,44))
__device__ const int g_H[LMAPS] = {32,32,32,32,32,32, 16,16,16,16,16,16};
__device__ const int g_W[LMAPS] = {88,88,88,88,88,88, 44,44,44,44,44,44};
__device__ const int g_S[LMAPS] = {0,2816,5632,8448,11264,14080,
                                   16896,17600,18304,19008,19712,20416};

// ---------------------------------------------------------------------------
// Generic GEMM: D[M x N] = A[M x K] (row-major f32) * W^T + bias (opt relu)
// W row-major [N x K].  128x64 tile / 256-thread workgroup, 8 waves,
// each wave: 16 rows x 64 cols (1 A-frag, 4 B-frags, 4 WMMA per k-step).
// ---------------------------------------------------------------------------
template <int RELU>
__global__ __launch_bounds__(256)
void k_gemm(const float* __restrict__ A, const float* __restrict__ W,
            const float* __restrict__ bias, float* __restrict__ D,
            int M, int N, int K)
{
  __shared__ __bf16 As[128][40];   // [row][k], 80B row stride (16B aligned)
  __shared__ __bf16 Bs[64][40];    // [col][k]

  const int tid  = threadIdx.x;
  const int row0 = blockIdx.x * 128;
  const int col0 = blockIdx.y * 64;
  const int lane = tid & 31;
  const int wv   = tid >> 5;       // 8 waves -> 16-row slices
  const int half = lane >> 4;
  const int mm   = lane & 15;

  f32x8 acc[4] = {{}, {}, {}, {}};

  for (int k0 = 0; k0 < K; k0 += 32) {
    // stage A tile 128x32: 4x float4 per thread, branch-free zero padding
#pragma unroll
    for (int i = 0; i < 4; ++i) {
      int e  = tid + i * 256;              // 0..1023
      int r  = e >> 3;                     // 0..127
      int c4 = (e & 7) * 4;
      int gr = row0 + r;
      int grc = gr < M ? gr : M - 1;
      float4 v = *(const float4*)(A + (long)grc * K + k0 + c4);
      bool ok = gr < M;
      bf16x4 b;
      b[0] = (__bf16)(ok ? v.x : 0.0f);
      b[1] = (__bf16)(ok ? v.y : 0.0f);
      b[2] = (__bf16)(ok ? v.z : 0.0f);
      b[3] = (__bf16)(ok ? v.w : 0.0f);
      *(bf16x4*)&As[r][c4] = b;
    }
    // stage B tile 64 cols x 32 k, stored [n][k]: contiguous both sides
#pragma unroll
    for (int i = 0; i < 2; ++i) {
      int e  = tid + i * 256;              // 0..511
      int n  = e >> 3;                     // 0..63
      int k4 = (e & 7) * 4;
      float4 v = *(const float4*)(W + (long)(col0 + n) * K + k0 + k4);
      bf16x4 b;
      b[0] = (__bf16)v.x; b[1] = (__bf16)v.y;
      b[2] = (__bf16)v.z; b[3] = (__bf16)v.w;
      *(bf16x4*)&Bs[n][k4] = b;
    }
    __syncthreads();

    // A frag: lane<16 -> K[0..15]; lane>=16 -> K[16..31]
    const int ar = wv * 16 + mm;
    bf16x16 af = ld_frag(&As[ar][half * 16], &As[ar][half * 16 + 8]);
#pragma unroll
    for (int t = 0; t < 4; ++t) {
      // B frag: lane = column, 16 contiguous K values per lane half
      const int bn = t * 16 + mm;
      bf16x16 bfr = ld_frag(&Bs[bn][half * 16], &Bs[bn][half * 16 + 8]);
      acc[t] = wmma_bf16f32(af, bfr, acc[t]);
    }
    __syncthreads();
  }

  // epilogue: D VGPR j -> row (j + 8*half), col = lane&15
#pragma unroll
  for (int t = 0; t < 4; ++t) {
    const int c  = col0 + t * 16 + mm;
    const float bv = bias[c];              // unconditional, hoisted per t
#pragma unroll
    for (int j = 0; j < 8; ++j) {
      int r = row0 + wv * 16 + half * 8 + j;
      float v = acc[t][j] + bv;
      if (RELU) v = fmaxf(v, 0.0f);
      if (r < M) D[(long)r * N + c] = v;
    }
  }
}

// ---------------------------------------------------------------------------
// build_value: value[t_base+t][o] = sum_c feat[nv][c][hw] * projW[o][c] + b[o]
// feat: (6, 256, HW); 64-row tiles never cross a camera boundary.
// 64x64 tile / workgroup; wave -> 16 rows x 32 cols (2 WMMA / k-step).
// ---------------------------------------------------------------------------
__global__ __launch_bounds__(256)
void k_build_value(const float* __restrict__ feat, const float* __restrict__ W,
                   const float* __restrict__ bias, float* __restrict__ value,
                   int t_base, int HW)
{
  __shared__ __bf16 As[64][40];
  __shared__ __bf16 Bs[64][40];

  const int tid  = threadIdx.x;
  const int tl0  = blockIdx.x * 64;       // local row within level
  const int col0 = blockIdx.y * 64;
  const int nv   = tl0 / HW;
  const int hw0  = tl0 % HW;
  const float* fb = feat + (long)nv * CDIM * HW + hw0;

  const int lane = tid & 31;
  const int wv   = tid >> 5;
  const int wr   = wv & 3;
  const int wc   = wv >> 2;
  const int half = lane >> 4;
  const int mm   = lane & 15;

  f32x8 acc0 = {};
  f32x8 acc1 = {};

  for (int k0 = 0; k0 < CDIM; k0 += 32) {
    // A[r][c] = fb[(k0+c)*HW + r]: float4 along r, scatter 4 rows into LDS
#pragma unroll
    for (int i = 0; i < 2; ++i) {
      int e  = tid + i * 256;              // 0..511
      int c  = e >> 4;                     // 0..31
      int r4 = (e & 15) * 4;               // 0,4,...,60
      float4 v = *(const float4*)(fb + (long)(k0 + c) * HW + r4);
      As[r4 + 0][c] = (__bf16)v.x;
      As[r4 + 1][c] = (__bf16)v.y;
      As[r4 + 2][c] = (__bf16)v.z;
      As[r4 + 3][c] = (__bf16)v.w;
    }
#pragma unroll
    for (int i = 0; i < 2; ++i) {
      int e  = tid + i * 256;
      int n  = e >> 3;
      int k4 = (e & 7) * 4;
      float4 v = *(const float4*)(W + (long)(col0 + n) * CDIM + k0 + k4);
      bf16x4 b;
      b[0] = (__bf16)v.x; b[1] = (__bf16)v.y;
      b[2] = (__bf16)v.z; b[3] = (__bf16)v.w;
      *(bf16x4*)&Bs[n][k4] = b;
    }
    __syncthreads();

    const int ar = wr * 16 + mm;
    bf16x16 af = ld_frag(&As[ar][half * 16], &As[ar][half * 16 + 8]);
    const int bn0 = wc * 32 + mm;
    bf16x16 bf0 = ld_frag(&Bs[bn0][half * 16],      &Bs[bn0][half * 16 + 8]);
    bf16x16 bf1 = ld_frag(&Bs[bn0 + 16][half * 16], &Bs[bn0 + 16][half * 16 + 8]);
    acc0 = wmma_bf16f32(af, bf0, acc0);
    acc1 = wmma_bf16f32(af, bf1, acc1);
    __syncthreads();
  }

  const int c = col0 + wc * 32 + mm;
  const float b0 = bias[c];
  const float b1 = bias[c + 16];
#pragma unroll
  for (int j = 0; j < 8; ++j) {
    int r = t_base + tl0 + wr * 16 + half * 8 + j;
    value[(long)r * CDIM + c]      = acc0[j] + b0;
    value[(long)r * CDIM + c + 16] = acc1[j] + b1;
  }
}

// ---------------------------------------------------------------------------
// reference points: project BEV grid -> ref (Nq, 12, 3, 2)
// ---------------------------------------------------------------------------
__global__ void k_ref(const float* __restrict__ l2i, float* __restrict__ ref)
{
  int idx = blockIdx.x * 256 + threadIdx.x;
  if (idx >= NQ * KDEP) return;
  int q  = idx / KDEP;
  int kd = idx % KDEP;
  float x = -49.5f + (float)(q % 100);
  float y = -49.5f + (float)(q / 100);
  float z = -6.0f + 4.0f * (float)kd;

  for (int cam = 0; cam < 6; ++cam) {
    const float* M = l2i + cam * 16;
    float c0 = M[0] * x + M[1] * y + M[2]  * z + M[3];
    float c1 = M[4] * x + M[5] * y + M[6]  * z + M[7];
    float c2 = M[8] * x + M[9] * y + M[10] * z + M[11];
    float zc = fmaxf(c2, 1e-5f);
    float u  = c0 / zc;
    float v  = c1 / zc;
    bool valid = zc > 1e-4f;
#pragma unroll
    for (int lvl = 0; lvl < 2; ++lvl) {
      float Wf = lvl ? 44.0f : 88.0f;
      float Hf = lvl ? 16.0f : 32.0f;
      float un = u / (Wf - 1.0f);
      float vn = v / (Hf - 1.0f);
      bool inb = valid && un >= 0.0f && un <= 1.0f && vn >= 0.0f && vn <= 1.0f;
      int  l   = lvl * 6 + cam;
      long o   = (((long)q * LMAPS + l) * KDEP + kd) * 2;
      ref[o]     = inb ? un : -2.0f;
      ref[o + 1] = inb ? vn : -2.0f;
    }
  }
}

// ---------------------------------------------------------------------------
// softmax over 48 (L*POINTS) per (q, head), in place on (Nq, 384)
// ---------------------------------------------------------------------------
__global__ void k_softmax(float* __restrict__ aw)
{
  int idx = blockIdx.x * 256 + threadIdx.x;
  if (idx >= NQ * NHEADS) return;
  float* p = aw + (long)(idx >> 3) * (NHEADS * 48) + (idx & 7) * 48;
  float mx = -3.4e38f;
  for (int i = 0; i < 48; ++i) mx = fmaxf(mx, p[i]);
  float s = 0.0f;
  for (int i = 0; i < 48; ++i) s += __expf(p[i] - mx);
  float inv = 1.0f / s;
  for (int i = 0; i < 48; ++i) p[i] = __expf(p[i] - mx) * inv;
}

// ---------------------------------------------------------------------------
// deformable sampler: one wave per (q, head), lane = head-dim channel.
// out[q][h*32+d] = sum_l sum_p aw/K * sum_k bilinear(v_l, ref+off)
// ---------------------------------------------------------------------------
__global__ __launch_bounds__(256)
void k_msda(const float* __restrict__ v, const float* __restrict__ ref,
            const float* __restrict__ off, const float* __restrict__ aw,
            float* __restrict__ out)
{
  int pair = blockIdx.x * 8 + (threadIdx.x >> 5);
  if (pair >= NQ * NHEADS) return;
  int q    = pair >> 3;
  int h    = pair & 7;
  int lane = threadIdx.x & 31;

  const float* refq = ref + (long)q * (LMAPS * KDEP * 2);
  const float* offq = off + (long)q * (NHEADS * LMAPS * NPTS * 2) + h * (LMAPS * NPTS * 2);
  const float* awq  = aw  + (long)q * (NHEADS * LMAPS * NPTS)     + h * (LMAPS * NPTS);

  float acc = 0.0f;
  for (int l = 0; l < LMAPS; ++l) {
    int Hh = g_H[l], Ww = g_W[l], s = g_S[l];
    const float* vb = v + (long)s * CDIM + h * HD + lane;
#pragma unroll
    for (int p = 0; p < NPTS; ++p) {
      float a  = awq[l * NPTS + p] * (1.0f / (float)KDEP);
      float ox = offq[(l * NPTS + p) * 2 + 0];
      float oy = offq[(l * NPTS + p) * 2 + 1];
#pragma unroll
      for (int kd = 0; kd < KDEP; ++kd) {
        float rx = refq[(l * KDEP + kd) * 2 + 0];
        float ry = refq[(l * KDEP + kd) * 2 + 1];
        // grid_sample(align_corners=False): pixel = loc*W - 0.5 (off already /W)
        float x = rx * (float)Ww + ox - 0.5f;
        float y = ry * (float)Hh + oy - 0.5f;
        float x0f = floorf(x), y0f = floorf(y);
        float fx = x - x0f, fy = y - y0f;
        int x0 = (int)x0f, y0 = (int)y0f;
        float smp = 0.0f;
#pragma unroll
        for (int dy = 0; dy < 2; ++dy) {
#pragma unroll
          for (int dx = 0; dx < 2; ++dx) {
            int xi = x0 + dx, yi = y0 + dy;
            if (xi >= 0 && xi < Ww && yi >= 0 && yi < Hh) {   // wave-uniform
              float wgt = (dx ? fx : 1.0f - fx) * (dy ? fy : 1.0f - fy);
              smp += wgt * vb[(long)(yi * Ww + xi) * CDIM];
            }
          }
        }
        acc += a * smp;
      }
    }
  }
  out[(long)q * CDIM + h * HD + lane] = acc;
}

// ---------------------------------------------------------------------------
// out = LayerNorm(a + b) * w + bias ; one wave per row of 256
// ---------------------------------------------------------------------------
__global__ __launch_bounds__(256)
void k_add_ln(const float* __restrict__ a, const float* __restrict__ b,
              const float* __restrict__ w, const float* __restrict__ bi,
              float* __restrict__ out, int rows)
{
  int row = blockIdx.x * 8 + (threadIdx.x >> 5);
  if (row >= rows) return;
  int lane = threadIdx.x & 31;
  float t[8];
  float s = 0.0f, s2 = 0.0f;
#pragma unroll
  for (int j = 0; j < 8; ++j) {
    int c = lane + j * 32;
    float v = a[(long)row * CDIM + c] + b[(long)row * CDIM + c];
    t[j] = v; s += v; s2 += v * v;
  }
#pragma unroll
  for (int m = 16; m > 0; m >>= 1) {
    s  += __shfl_xor(s,  m);
    s2 += __shfl_xor(s2, m);
  }
  float mean = s * (1.0f / 256.0f);
  float var  = s2 * (1.0f / 256.0f) - mean * mean;
  float inv  = rsqrtf(var + 1e-5f);
#pragma unroll
  for (int j = 0; j < 8; ++j) {
    int c = lane + j * 32;
    out[(long)row * CDIM + c] = (t[j] - mean) * inv * w[c] + bi[c];
  }
}

// ---------------------------------------------------------------------------
// final transpose: (Nq, C) -> (C, 100, 100)
// ---------------------------------------------------------------------------
__global__ void k_transpose(const float* __restrict__ q, float* __restrict__ out)
{
  int idx = blockIdx.x * 256 + threadIdx.x;
  if (idx >= CDIM * NQ) return;
  int c  = idx / NQ;
  int qi = idx % NQ;
  out[idx] = q[(long)qi * CDIM + c];
}

// ---------------------------------------------------------------------------
// host launcher
// ---------------------------------------------------------------------------
extern "C" void kernel_launch(void* const* d_in, const int* in_sizes, int n_in,
                              void* d_out, int out_size, void* d_ws, size_t ws_size,
                              hipStream_t stream)
{
  (void)in_sizes; (void)n_in; (void)out_size; (void)ws_size;

  // input order: feat0, feat1, lidar2img, bev_embed, proj0_w/b, proj1_w/b,
  // then per block: off_w/b, aw_w/b, vproj_w/b, oproj_w/b, ln1_w/b,
  //                 ffn1_w/b, ffn2_w/b, ln2_w/b  (16 tensors / block)
  const float* feat0 = (const float*)d_in[0];
  const float* feat1 = (const float*)d_in[1];
  const float* l2i   = (const float*)d_in[2];
  const float* bev   = (const float*)d_in[3];
  const float* p0w   = (const float*)d_in[4];
  const float* p0b   = (const float*)d_in[5];
  const float* p1w   = (const float*)d_in[6];
  const float* p1b   = (const float*)d_in[7];

  float* ws = (float*)d_ws;
  // workspace layout (floats)
  float* VALUE = ws;                         // 21120*256 = 5,406,720
  float* REF   = VALUE + 5406720;            //   720,000
  float* VPROJ = REF   + 720000;             // 5,406,720
  float* QA    = VPROJ + 5406720;            // 2,560,000
  float* QB    = QA    + 2560000;            // 2,560,000
  float* OFF   = QB    + 2560000;            // 7,680,000  (attn + ffn1 reuse)
  float* AW    = OFF   + 7680000;            // 3,840,000  (x reuse)
  float* MSDA  = AW    + 3840000;            // 2,560,000  (ffn2 reuse)
  float* ATTN  = OFF;                        // alias (off dead after sampling)
  float* FFN1  = OFF + 2560000;              // alias
  float* XBUF  = AW;                         // alias (aw dead after sampling)
  float* FFN2  = MSDA;                       // alias (msda dead after oproj)

  // geometry
  k_ref<<<(NQ * KDEP + 255) / 256, 256, 0, stream>>>(l2i, REF);

  // value = channel-linear over both levels
  k_build_value<<<dim3(16896 / 64, 4), 256, 0, stream>>>(feat0, p0w, p0b, VALUE, 0,     32 * 88);
  k_build_value<<<dim3( 4224 / 64, 4), 256, 0, stream>>>(feat1, p1w, p1b, VALUE, 16896, 16 * 44);

  const int MQ_BLKS = (NQ + 127) / 128;   // 79
  const int MT_BLKS = TTOT / 128;         // 165

  const float* q_in = bev;
  float* q_out = QA;
  for (int blk = 0; blk < 2; ++blk) {
    const int base = 8 + 16 * blk;
    const float* offw = (const float*)d_in[base + 0];
    const float* offb = (const float*)d_in[base + 1];
    const float* aww  = (const float*)d_in[base + 2];
    const float* awb  = (const float*)d_in[base + 3];
    const float* vpw  = (const float*)d_in[base + 4];
    const float* vpb  = (const float*)d_in[base + 5];
    const float* opw  = (const float*)d_in[base + 6];
    const float* opb  = (const float*)d_in[base + 7];
    const float* ln1w = (const float*)d_in[base + 8];
    const float* ln1b = (const float*)d_in[base + 9];
    const float* f1w  = (const float*)d_in[base + 10];
    const float* f1b  = (const float*)d_in[base + 11];
    const float* f2w  = (const float*)d_in[base + 12];
    const float* f2b  = (const float*)d_in[base + 13];
    const float* ln2w = (const float*)d_in[base + 14];
    const float* ln2b = (const float*)d_in[base + 15];

    // v = value @ vproj_w^T + b      (21120 x 256)
    k_gemm<0><<<dim3(MT_BLKS, 4), 256, 0, stream>>>(VALUE, vpw, vpb, VPROJ, TTOT, 256, 256);
    // off = q @ off_w^T + b          (10000 x 768)
    k_gemm<0><<<dim3(MQ_BLKS, 12), 256, 0, stream>>>(q_in, offw, offb, OFF, NQ, 768, 256);
    // aw_raw = q @ aw_w^T + b        (10000 x 384)
    k_gemm<0><<<dim3(MQ_BLKS, 6), 256, 0, stream>>>(q_in, aww, awb, AW, NQ, 384, 256);
    // softmax over 48 per (q, head)
    k_softmax<<<(NQ * NHEADS + 255) / 256, 256, 0, stream>>>(AW);
    // deformable bilinear gather
    k_msda<<<NQ, 256, 0, stream>>>(VPROJ, REF, OFF, AW, MSDA);
    // attn = msda @ oproj_w^T + b    (10000 x 256)
    k_gemm<0><<<dim3(MQ_BLKS, 4), 256, 0, stream>>>(MSDA, opw, opb, ATTN, NQ, 256, 256);
    // x = LN(q + attn)
    k_add_ln<<<(NQ + 7) / 8, 256, 0, stream>>>(q_in, ATTN, ln1w, ln1b, XBUF, NQ);
    // ffn1 = relu(x @ ffn1_w^T + b)  (10000 x 512)
    k_gemm<1><<<dim3(MQ_BLKS, 8), 256, 0, stream>>>(XBUF, f1w, f1b, FFN1, NQ, 512, 256);
    // ffn2 = ffn1 @ ffn2_w^T + b     (10000 x 256)
    k_gemm<0><<<dim3(MQ_BLKS, 4), 256, 0, stream>>>(FFN1, f2w, f2b, FFN2, NQ, 256, 512);
    // q_next = LN(x + ffn2)
    k_add_ln<<<(NQ + 7) / 8, 256, 0, stream>>>(XBUF, FFN2, ln2w, ln2b, q_out, NQ);

    q_in  = q_out;
    q_out = QB;
  }

  // (Nq, C) -> (C, 100, 100)
  k_transpose<<<(CDIM * NQ + 255) / 256, 256, 0, stream>>>(q_in, (float*)d_out);
}
